// FusionGAT2_1906965479908
// MI455X (gfx1250) — matrix operand order repaired and checked by reading, not compiled, not run on hardware
//
#include <hip/hip_runtime.h>
#include <hip/hip_bf16.h>
#include <cstddef>

// ---------------------------------------------------------------------------
// CDNA5 (gfx1250) fused multi-graph GAT.
// Matrix path: v_wmma_f32_16x16x32_f16 (f16 A/B, f32 accumulate).
// All WMMA B-operands are pre-packed into fragment order so the inner loops
// are pure b128 loads + wmma (no guards, no element inserts).
// Attention is flash-style: the NxN score matrix is never materialized.
// ---------------------------------------------------------------------------

typedef _Float16 v16h __attribute__((ext_vector_type(16)));
typedef _Float16 v8h  __attribute__((ext_vector_type(8)));
typedef float    v8f  __attribute__((ext_vector_type(8)));

#define LRELU_ALPHA 0.2f

static __device__ __forceinline__ v16h cat8(v8h lo, v8h hi) {
  return __builtin_shufflevector(lo, hi, 0, 1, 2, 3, 4, 5, 6, 7,
                                 8, 9, 10, 11, 12, 13, 14, 15);
}

// ---------------------------------------------------------------------------
// f32 -> f16 row-major conversion (for A matrices)
// ---------------------------------------------------------------------------
__global__ void f32_to_f16_kernel(const float* __restrict__ src,
                                  _Float16* __restrict__ dst, int nElem) {
  int i = blockIdx.x * blockDim.x + threadIdx.x;
  if (i < nElem) dst[i] = (_Float16)src[i];
}

// ---------------------------------------------------------------------------
// Pack a batch of row-major f32 B matrices [nb][K x N] into WMMA fragment
// order f16: Bp[b][kt][nt][lane][e] with
//   lane = half*16 + m,  value = B[kt*32 + half*16 + e][nt*16 + m]
// Columns beyond N are zero-filled (pads N=8 -> 16).
// ---------------------------------------------------------------------------
__global__ void pack_b_kernel(const float* __restrict__ B,
                              _Float16* __restrict__ Bp,
                              int nb, int K, int N, int Ntiles) {
  int idx = blockIdx.x * 256 + threadIdx.x;
  int per = (K >> 5) * Ntiles * 512;
  if (idx >= nb * per) return;
  int b    = idx / per;
  int rem  = idx % per;
  int e    = rem & 15;
  int lane = (rem >> 4) & 31;
  int tile = rem >> 9;             // kt*Ntiles + nt
  int nt   = tile % Ntiles;
  int kt   = tile / Ntiles;
  int m = lane & 15, half = lane >> 4;
  int col = nt * 16 + m;
  int k   = kt * 32 + half * 16 + e;
  float v = (col < N) ? B[((size_t)b * K + k) * N + col] : 0.f;
  Bp[idx] = (_Float16)v;
}

__global__ void pad_bias_kernel(const float* __restrict__ b,
                                float* __restrict__ bp, int n, int total) {
  int i = threadIdx.x;
  if (i < total) bp[i] = (i < n) ? b[i] : 0.f;
}

// ---------------------------------------------------------------------------
// WMMA GEMM, one wave per 16x(16*NST) tile.
//   A  : f16 row-major [M x K], K multiple of 32 (two b128 loads / k-tile)
//   Bp : fragment-packed (see pack_b_kernel), Ntiles_total = gridDim.y*NST
//   Cf : optional f32 out (guarded by colB < N), ldC/colOff
//   Ch : optional f16 row-major out (full tiles only)
//   Cp : optional f16 fragment-packed out (for use as next B; gridDim.y==1)
// act: 0 = none, 1 = ELU.
// A-frag: element e -> K = half*16+e, row = m
// B-frag: element e -> K = half*16+e, col = m
// C/D   : VGPR j -> row = j + half*8, col = m
// ---------------------------------------------------------------------------
template <int NST>
__global__ void gemm_wmma(const _Float16* __restrict__ A,
                          const _Float16* __restrict__ Bp,
                          const float* __restrict__ bias,
                          float* __restrict__ Cf,
                          _Float16* __restrict__ Ch,
                          _Float16* __restrict__ Cp,
                          int N, int K, int ldA, int ldC, int colOff, int act) {
  int lane = threadIdx.x;
  int m    = lane & 15;
  int half = lane >> 4;
  int row0 = blockIdx.x << 4;
  int ntTot = gridDim.y * NST;
  int nt0   = blockIdx.y * NST;

  v8f acc[NST] = {};
  const _Float16* Arow = A + (size_t)(row0 + m) * ldA;

  for (int k0 = 0; k0 < K; k0 += 32) {
    v8h alo = *(const v8h*)(Arow + k0 + half * 16);
    v8h ahi = *(const v8h*)(Arow + k0 + half * 16 + 8);
    v16h aF = cat8(alo, ahi);
    int kt = k0 >> 5;
    const _Float16* bpt =
        Bp + ((size_t)(kt * ntTot + nt0) * 32 + lane) * 16;
#pragma unroll
    for (int s = 0; s < NST; ++s) {
      v8h blo = *(const v8h*)(bpt + s * 512);
      v8h bhi = *(const v8h*)(bpt + s * 512 + 8);
      v16h bF = cat8(blo, bhi);
      acc[s] = __builtin_amdgcn_wmma_f32_16x16x32_f16(false, aF, false, bF,
                                                      (short)0, acc[s],
                                                      false, false);
    }
  }

#pragma unroll
  for (int s = 0; s < NST; ++s) {
    int colB = (nt0 + s) * 16 + m;
    bool colOK = colB < N;
    float bv = (bias != nullptr) ? bias[colB] : 0.f;  // bias buffers padded
#pragma unroll
    for (int j = 0; j < 8; ++j) {
      int r = row0 + j + half * 8;
      float v = acc[s][j] + bv;
      if (act == 1) v = v > 0.f ? v : (__expf(v) - 1.f);  // ELU
      size_t idx = (size_t)r * ldC + colOff + colB;
      if (Cf != nullptr && colOK) Cf[idx] = v;
      if (Ch != nullptr) Ch[idx] = (_Float16)v;
      if (Cp != nullptr) {
        // scatter into fragment-packed order: this value is B[k=r][col=colB]
        size_t pidx = ((size_t)(r >> 5) * NST + s) * 512 +
                      (size_t)((((r >> 4) & 1) << 4) | m) * 16 + (r & 15);
        Cp[pidx] = (_Float16)v;
      }
    }
  }
}

// ---------------------------------------------------------------------------
// Attention projections: fsrc[h,n] = Wh[h,n,:].a[h,0,:], fdst = .a[h,1,:]
// ---------------------------------------------------------------------------
__global__ void att_scores_kernel(const _Float16* __restrict__ Whh,
                                  const float* __restrict__ a,
                                  float* __restrict__ fsrc,
                                  float* __restrict__ fdst, int n, int O) {
  int idx = blockIdx.x * blockDim.x + threadIdx.x;  // H*n threads, n=4096
  int h = idx >> 12;
  int r = idx & 4095;
  if (h >= 4) return;
  const _Float16* w  = Whh + ((size_t)h * n + r) * O;
  const float*    a0 = a + (size_t)h * 2 * O;
  const float*    a1 = a0 + O;
  float s0 = 0.f, s1 = 0.f;
  for (int o = 0; o < O; o += 8) {
    v8h wv = *(const v8h*)(w + o);
#pragma unroll
    for (int e = 0; e < 8; ++e) {
      float f = (float)wv[e];
      s0 += f * a0[o + e];
      s1 += f * a1[o + e];
    }
  }
  fsrc[h * n + r] = s0;
  fdst[h * n + r] = s1;
}

// ---------------------------------------------------------------------------
// Flash pass 1: per-row masked softmax stats of e = lrelu(fsrc[r]+fdst[c]).
// Rank-1 structure => no matmul. One block per row, online max/sum per head.
// Rows are never empty (self loops) so S>0; -3e38 sentinel keeps exp finite.
// ---------------------------------------------------------------------------
__global__ void softmax_stats_kernel(const float* __restrict__ adjI,
                                     const float* __restrict__ fsrc,
                                     const float* __restrict__ fdst,
                                     float* __restrict__ mOut,
                                     float* __restrict__ sOut, int n) {
  __shared__ float sm[4][256];
  __shared__ float ss[4][256];
  int row = blockIdx.x;
  int tid = threadIdx.x;
  float fr[4], mh[4], sh[4];
#pragma unroll
  for (int h = 0; h < 4; ++h) {
    fr[h] = fsrc[h * n + row];
    mh[h] = -3.0e38f;
    sh[h] = 0.f;
  }
  const float* arow = adjI + (size_t)row * n;
  for (int c = tid; c < n; c += 256) {
    float av = arow[c];
    if (av > 0.f) {
#pragma unroll
      for (int h = 0; h < 4; ++h) {
        float t = fr[h] + fdst[h * n + c];
        t = t > 0.f ? t : LRELU_ALPHA * t;
        float nm = fmaxf(mh[h], t);
        sh[h] = sh[h] * __expf(mh[h] - nm) + __expf(t - nm);
        mh[h] = nm;
      }
    }
  }
#pragma unroll
  for (int h = 0; h < 4; ++h) { sm[h][tid] = mh[h]; ss[h][tid] = sh[h]; }
  __syncthreads();
  for (int st = 128; st > 0; st >>= 1) {
    if (tid < st) {
#pragma unroll
      for (int h = 0; h < 4; ++h) {
        float m1 = sm[h][tid], m2 = sm[h][tid + st];
        float M = fmaxf(m1, m2);
        float s = ss[h][tid] * __expf(m1 - M) + ss[h][tid + st] * __expf(m2 - M);
        sm[h][tid] = M;
        ss[h][tid] = s;
      }
    }
    __syncthreads();
  }
  if (tid < 4) {
    mOut[tid * n + row] = sm[tid][0];
    sOut[tid * n + row] = ss[tid][0];
  }
}

// ---------------------------------------------------------------------------
// Flash pass 2: out[r,:] = ELU( sum_c P[r,c] * Wh[c,:] ), 16-row tile / wave.
// P is regenerated per 16x32 tile directly into the WMMA A-fragment (f16);
// Wh comes from its fragment-packed mirror (two b128 loads per strip).
// NST = O/16 (4 for o=64, 2 for o=32). Writes the head's slice of the
// concatenated f16 activation matrix (row-major, consumed as next A).
// ---------------------------------------------------------------------------
template <int NST>
__global__ void gat_aggregate_kernel(const float* __restrict__ adjI,
                                     const _Float16* __restrict__ Whp,
                                     const float* __restrict__ fsrc,
                                     const float* __restrict__ fdst,
                                     const float* __restrict__ mrow,
                                     const float* __restrict__ srow,
                                     _Float16* __restrict__ outcat,
                                     int n, int ldOut) {
  constexpr int O = NST * 16;
  int h    = blockIdx.y;
  int row0 = blockIdx.x << 4;
  int lane = threadIdx.x;
  int m    = lane & 15;
  int half = lane >> 4;
  int r    = row0 + m;

  float fr   = fsrc[h * n + r];
  float mr   = mrow[h * n + r];
  float invS = 1.0f / srow[h * n + r];

  v8f acc[NST] = {};
  const float*    arow = adjI + (size_t)r * n;
  const float*    fdh  = fdst + (size_t)h * n;
  const _Float16* whp  = Whp + (size_t)h * n * O;   // packed per head

  for (int c0 = 0; c0 < n; c0 += 32) {
    v16h aF;
#pragma unroll
    for (int g = 0; g < 2; ++g) {  // g=0 -> elems 0..7 (K=half*16+e), g=1 -> +8
      int cb = c0 + half * 16 + g * 8;
      const float4 A0 = *(const float4*)(arow + cb);
      const float4 A1 = *(const float4*)(arow + cb + 4);
      const float4 F0 = *(const float4*)(fdh + cb);
      const float4 F1 = *(const float4*)(fdh + cb + 4);
      float av[8] = {A0.x, A0.y, A0.z, A0.w, A1.x, A1.y, A1.z, A1.w};
      float fv[8] = {F0.x, F0.y, F0.z, F0.w, F1.x, F1.y, F1.z, F1.w};
#pragma unroll
      for (int e = 0; e < 8; ++e) {
        float t = fr + fv[e];
        t = t > 0.f ? t : LRELU_ALPHA * t;
        float p = (av[e] > 0.f) ? __expf(t - mr) * invS : 0.f;
        aF[g * 8 + e] = (_Float16)p;
      }
    }
    int kt = c0 >> 5;
    const _Float16* bpt = whp + ((size_t)(kt * NST) * 32 + lane) * 16;
#pragma unroll
    for (int s = 0; s < NST; ++s) {
      v8h blo = *(const v8h*)(bpt + s * 512);
      v8h bhi = *(const v8h*)(bpt + s * 512 + 8);
      v16h bF = cat8(blo, bhi);
      acc[s] = __builtin_amdgcn_wmma_f32_16x16x32_f16(false, aF, false, bF,
                                                      (short)0, acc[s],
                                                      false, false);
    }
  }

#pragma unroll
  for (int s = 0; s < NST; ++s) {
#pragma unroll
    for (int j = 0; j < 8; ++j) {
      int rr = row0 + j + half * 8;
      float v = acc[s][j];
      v = v > 0.f ? v : (__expf(v) - 1.0f);  // ELU
      outcat[(size_t)rr * ldOut + h * O + s * 16 + m] = (_Float16)v;
    }
  }
}

// ---------------------------------------------------------------------------
// Final head: logits = cat16 @ Wf + bf ; log_softmax per row.
// ---------------------------------------------------------------------------
__global__ void final_head_kernel(const float* __restrict__ catF,
                                  const float* __restrict__ Wf,
                                  const float* __restrict__ bf,
                                  float* __restrict__ out, int n) {
  int r = blockIdx.x * blockDim.x + threadIdx.x;
  if (r >= n) return;
  float c[16];
#pragma unroll
  for (int k = 0; k < 16; ++k) c[k] = catF[(size_t)r * 16 + k];
  float lg[8];
#pragma unroll
  for (int j = 0; j < 8; ++j) {
    float s = bf[j];
#pragma unroll
    for (int k = 0; k < 16; ++k) s += c[k] * Wf[k * 8 + j];
    lg[j] = s;
  }
  float mx = lg[0];
#pragma unroll
  for (int j = 1; j < 8; ++j) mx = fmaxf(mx, lg[j]);
  float se = 0.f;
#pragma unroll
  for (int j = 0; j < 8; ++j) se += __expf(lg[j] - mx);
  float lse = mx + __logf(se);
#pragma unroll
  for (int j = 0; j < 8; ++j) out[(size_t)r * 8 + j] = lg[j] - lse;
}

__global__ void l1_kernel(const float* __restrict__ Wf,
                          float* __restrict__ out, int count) {
  if (threadIdx.x == 0 && blockIdx.x == 0) {
    float s = 0.f;
    for (int i = 0; i < count; ++i) s += fabsf(Wf[i]);
    out[0] = s / (float)count;
  }
}

// ---------------------------------------------------------------------------
// Host orchestration (graph-capture safe: launches only).
// ---------------------------------------------------------------------------
extern "C" void kernel_launch(void* const* d_in, const int* in_sizes, int n_in,
                              void* d_out, int out_size, void* d_ws,
                              size_t ws_size, hipStream_t stream) {
  (void)in_sizes; (void)n_in; (void)out_size; (void)ws_size;
  constexpr int NN = 4096, NF = 256, H1 = 64, H2 = 32, NCL = 8, NH = 4;

  const float* x   = (const float*)d_in[0];
  const float* adj = (const float*)d_in[1];
  const float* W1  = (const float*)d_in[2];
  const float* a1  = (const float*)d_in[3];
  const float* W2  = (const float*)d_in[4];
  const float* a2  = (const float*)d_in[5];
  const float* Wi1 = (const float*)d_in[6];
  const float* bi1 = (const float*)d_in[7];
  const float* Wi2 = (const float*)d_in[8];
  const float* bi2 = (const float*)d_in[9];
  const float* Wf  = (const float*)d_in[10];
  const float* bf  = (const float*)d_in[11];
  float* out = (float*)d_out;

  // ---- workspace carve-out (~12.5 MB), 256B aligned chunks ----
  char*  base = (char*)d_ws;
  size_t off  = 0;
  auto bump = [&](size_t bytes) -> char* {
    char* p = base + off;
    off += (bytes + 255) & ~(size_t)255;
    return p;
  };
  _Float16* xh    = (_Float16*)bump((size_t)NN * NF * 2);
  _Float16* W1p   = (_Float16*)bump((size_t)2 * NH * NF * H1 * 2);    // packed
  _Float16* W2p   = (_Float16*)bump((size_t)2 * NH * H1 * H2 * 2);    // packed
  _Float16* Wi1p  = (_Float16*)bump((size_t)NF * H1 * 2);             // packed
  _Float16* Wi2p  = (_Float16*)bump((size_t)(NH * H2) * 16 * 2);      // packed, padded
  float*    bi2p  = (float*)bump(16 * 4);
  _Float16* Wh1h  = (_Float16*)bump((size_t)NH * NN * H1 * 2);  // row-major
  _Float16* Wh1p  = (_Float16*)bump((size_t)NH * NN * H1 * 2);  // packed
  _Float16* Wh2h  = (_Float16*)bump((size_t)NH * NN * H2 * 2);
  _Float16* Wh2p  = (_Float16*)bump((size_t)NH * NN * H2 * 2);
  float* fsrc1 = (float*)bump((size_t)NH * NN * 4);
  float* fdst1 = (float*)bump((size_t)NH * NN * 4);
  float* m1    = (float*)bump((size_t)NH * NN * 4);
  float* s1    = (float*)bump((size_t)NH * NN * 4);
  float* fsrc2 = (float*)bump((size_t)NH * NN * 4);
  float* fdst2 = (float*)bump((size_t)NH * NN * 4);
  float* m2    = (float*)bump((size_t)NH * NN * 4);
  float* s2    = (float*)bump((size_t)NH * NN * 4);
  _Float16* xcat1h = (_Float16*)bump((size_t)NN * (NH * H1) * 2);
  _Float16* x2h    = (_Float16*)bump((size_t)NN * H1 * 2);
  _Float16* xcat2h = (_Float16*)bump((size_t)NN * (NH * H2) * 2);
  float* catF = (float*)bump((size_t)NN * (2 * NCL) * 4);

  // ---- one-time conversions / weight packing ----
  f32_to_f16_kernel<<<(NN * NF + 255) / 256, 256, 0, stream>>>(x, xh, NN * NF);
  {
    int nW1 = 2 * NH * NF * H1;     // 8 matrices of 256x64, Ntiles=4
    pack_b_kernel<<<(nW1 + 255) / 256, 256, 0, stream>>>(W1, W1p, 2 * NH, NF, H1, H1 / 16);
    int nW2 = 2 * NH * H1 * H2;     // 8 matrices of 64x32, Ntiles=2
    pack_b_kernel<<<(nW2 + 255) / 256, 256, 0, stream>>>(W2, W2p, 2 * NH, H1, H2, H2 / 16);
    int nWi1 = NF * H1;             // 256x64, Ntiles=4
    pack_b_kernel<<<(nWi1 + 255) / 256, 256, 0, stream>>>(Wi1, Wi1p, 1, NF, H1, H1 / 16);
    int nWi2 = (NH * H2) * 16;      // 128x8 padded to 16 cols, Ntiles=1
    pack_b_kernel<<<(nWi2 + 255) / 256, 256, 0, stream>>>(Wi2, Wi2p, 1, NH * H2, NCL, 1);
    pad_bias_kernel<<<1, 16, 0, stream>>>(bi2, bi2p, NCL, 16);
  }

  for (int i = 0; i < 2; ++i) {
    const float* adjI = adj + (size_t)i * NN * NN;

    // ---- layer 1: per-head feature GEMM (4096x256x64) ----
    for (int h = 0; h < NH; ++h)
      gemm_wmma<4><<<dim3(NN / 16, 1), 32, 0, stream>>>(
          xh, W1p + (size_t)(i * NH + h) * NF * H1, nullptr,
          nullptr, Wh1h + (size_t)h * NN * H1, Wh1p + (size_t)h * NN * H1,
          H1, NF, NF, H1, 0, 0);
    att_scores_kernel<<<(NH * NN) / 256, 256, 0, stream>>>(
        Wh1h, a1 + (size_t)i * NH * 2 * H1, fsrc1, fdst1, NN, H1);
    softmax_stats_kernel<<<NN, 256, 0, stream>>>(adjI, fsrc1, fdst1, m1, s1, NN);
    gat_aggregate_kernel<4><<<dim3(NN / 16, NH), 32, 0, stream>>>(
        adjI, Wh1p, fsrc1, fdst1, m1, s1, xcat1h, NN, NH * H1);
    // x2 = ELU(cat1 @ Wi1 + bi1)  (4096x256x64)
    gemm_wmma<4><<<dim3(NN / 16, 1), 32, 0, stream>>>(
        xcat1h, Wi1p, bi1, nullptr, x2h, nullptr,
        H1, NH * H1, NH * H1, H1, 0, 1);

    // ---- layer 2: per-head feature GEMM (4096x64x32) ----
    for (int h = 0; h < NH; ++h)
      gemm_wmma<2><<<dim3(NN / 16, 1), 32, 0, stream>>>(
          x2h, W2p + (size_t)(i * NH + h) * H1 * H2, nullptr,
          nullptr, Wh2h + (size_t)h * NN * H2, Wh2p + (size_t)h * NN * H2,
          H2, H1, H1, H2, 0, 0);
    att_scores_kernel<<<(NH * NN) / 256, 256, 0, stream>>>(
        Wh2h, a2 + (size_t)i * NH * 2 * H2, fsrc2, fdst2, NN, H2);
    softmax_stats_kernel<<<NN, 256, 0, stream>>>(adjI, fsrc2, fdst2, m2, s2, NN);
    gat_aggregate_kernel<2><<<dim3(NN / 16, NH), 32, 0, stream>>>(
        adjI, Wh2p, fsrc2, fdst2, m2, s2, xcat2h, NN, NH * H2);
    // catF[:, i*8 .. i*8+7] = ELU(cat2 @ Wi2 + bi2)  (4096x128x8, store-guarded)
    gemm_wmma<1><<<dim3(NN / 16, 1), 32, 0, stream>>>(
        xcat2h, Wi2p, bi2p, catF, nullptr, nullptr,
        NCL, NH * H2, NH * H2, 2 * NCL, i * NCL, 1);
  }

  final_head_kernel<<<NN / 256, 256, 0, stream>>>(catF, Wf, bf, out, NN);
  l1_kernel<<<1, 32, 0, stream>>>(Wf, out + (size_t)NN * NCL, 2 * NCL * NCL);
}
